// MoE_4818953306216
// MI455X (gfx1250) — matrix operand order repaired and checked by reading, not compiled, not run on hardware
//
#include <hip/hip_runtime.h>
#include <hip/hip_bf16.h>
#include <math.h>

#define B_  4
#define S_  4096
#define D_  1024
#define H_  4096
#define RH_ 1024
#define E_  16
#define K_  2
#define T_  (B_ * S_)

#define BM 32                             /* rows per GEMM block (2 WMMA row-groups) */
#define MAXTILES ((T_ * K_) / BM + E_)    /* 1040 routed tiles */
#define PADCAP   (MAXTILES * BM)          /* 33280 padded assignment slots */

#define XLD 1032   /* bf16 LDS stride, x-tile rows: 2064B, 16B aligned, conflict-free */
#define HLD 136    /* bf16 LDS stride, act-tile rows: 272B, 16B aligned */
#define SMEM_BYTES ((BM * XLD + BM * HLD) * 2)   /* 74752 B dynamic LDS */

typedef __attribute__((ext_vector_type(16))) __bf16 v16bf;
typedef __attribute__((ext_vector_type(8)))  float  v8f;

static __device__ inline unsigned short f2bfu(float f) {
  unsigned u = __builtin_bit_cast(unsigned, f);
  u += 0x7FFFu + ((u >> 16) & 1u);            // round-to-nearest-even
  return (unsigned short)(u >> 16);
}
static __device__ inline __bf16 f2bf(float f) {
  unsigned short h = f2bfu(f);
  return __builtin_bit_cast(__bf16, h);
}
static __device__ inline v8f vzero8() {
  v8f v;
#pragma unroll
  for (int i = 0; i < 8; ++i) v[i] = 0.f;
  return v;
}

// A fragment: 16x32 bf16 from LDS (row-major, stride ldw) at column k0.
static __device__ inline v16bf load_a_lds(const __bf16* lds, int ldw, int k0, int lane) {
  const int m   = lane & 15;
  const int khi = (lane >> 4) << 3;
  const __bf16* row = lds + m * ldw + k0 + khi;
  union { v16bf v; uint4 q[2]; } a;
  a.q[0] = *(const uint4*)(row);
  a.q[1] = *(const uint4*)(row + 16);
  return a.v;
}

// B fragment from pre-swizzled bf16 weights (lane-vectors contiguous: 1KB/wave).
static __device__ inline v16bf load_b_sw(const __bf16* __restrict__ g, int Ktiles,
                                         int kt, int nt, int lane) {
  const __bf16* p = g + ((((size_t)nt * Ktiles + kt) << 5) + lane) * 16;
  union { v16bf v; uint4 q[2]; } b;
  b.q[0] = *(const uint4*)(p);
  b.q[1] = *(const uint4*)(p + 8);
  return b.v;
}

// ---------------------------------------------------------------------------
// 0a) fp32 -> bf16 (x; A-side consumed row-major via LDS)
// ---------------------------------------------------------------------------
__global__ __launch_bounds__(256)
void cvt_kernel(const float* __restrict__ src, __bf16* __restrict__ dst, int n4) {
  const int i = blockIdx.x * blockDim.x + threadIdx.x;
  if (i >= n4) return;
  const float4 v = ((const float4*)src)[i];
  uint2 p;
  p.x = (unsigned)f2bfu(v.x) | ((unsigned)f2bfu(v.y) << 16);
  p.y = (unsigned)f2bfu(v.z) | ((unsigned)f2bfu(v.w) << 16);
  ((uint2*)dst)[i] = p;
}

// ---------------------------------------------------------------------------
// 0b) fp32 [nmat][K][N] -> bf16 WMMA-B-swizzled [nmat][N/16][K/32][32][16]
// ---------------------------------------------------------------------------
__global__ __launch_bounds__(256)
void swz_kernel(const float* __restrict__ src, __bf16* __restrict__ dst,
                int K, int N, int nmat) {
  const int K32 = K >> 5, N16 = N >> 4;
  const size_t tot = (size_t)K32 * N16 * 32 * nmat;
  const size_t i = (size_t)blockIdx.x * blockDim.x + threadIdx.x;
  if (i >= tot) return;
  const int lane = (int)(i & 31);
  size_t t = i >> 5;
  const int kt = (int)(t % K32); t /= K32;
  const int nt = (int)(t % N16); t /= N16;
  const int m  = (int)t;
  const int n     = (nt << 4) + (lane & 15);
  const int kbase = (kt << 5) + ((lane >> 4) << 3);
  const float* s = src + ((size_t)m * K + kbase) * N + n;
  union { v16bf v; uint4 q[2]; } b;
#pragma unroll
  for (int j = 0; j < 16; ++j) b.v[j] = f2bf(s[(size_t)(j + ((j >> 3) << 3)) * N]);
  __bf16* d = dst + i * 16;
  *(uint4*)(d)     = b.q[0];
  *(uint4*)(d + 8) = b.q[1];
}

// ---------------------------------------------------------------------------
// 1) Router
// ---------------------------------------------------------------------------
__global__ __launch_bounds__(256)
void router_kernel(const float* __restrict__ x, const float* __restrict__ rw,
                   const float* __restrict__ ebias,
                   int* __restrict__ top_idx, float* __restrict__ top_gate,
                   int* __restrict__ cnt) {
  const int wave = threadIdx.x >> 5;
  const int lane = threadIdx.x & 31;
  const int t = blockIdx.x * 8 + wave;
  float acc[E_];
#pragma unroll
  for (int e = 0; e < E_; ++e) acc[e] = 0.f;
  const float* xr = x + (size_t)t * D_;
  for (int d = lane; d < D_; d += 32) {
    const float xv = xr[d];
    const float* w = rw + d * E_;
#pragma unroll
    for (int e = 0; e < E_; ++e) acc[e] = fmaf(xv, w[e], acc[e]);
  }
#pragma unroll
  for (int e = 0; e < E_; ++e) {
#pragma unroll
    for (int m = 16; m >= 1; m >>= 1) acc[e] += __shfl_xor(acc[e], m, 32);
  }
  if (lane == 0) {
    float score[E_], sel[E_];
#pragma unroll
    for (int e = 0; e < E_; ++e) {
      score[e] = __builtin_amdgcn_rcpf(1.f + __expf(-acc[e]));
      sel[e] = score[e] + ebias[e];
    }
    int i0 = 0;
#pragma unroll
    for (int e = 1; e < E_; ++e) if (sel[e] > sel[i0]) i0 = e;   // ties -> lowest idx
    int i1 = (i0 == 0) ? 1 : 0;
#pragma unroll
    for (int e = 0; e < E_; ++e) if (e != i0 && sel[e] > sel[i1]) i1 = e;
    top_idx[t * 2 + 0] = i0;  top_gate[t * 2 + 0] = score[i0];
    top_idx[t * 2 + 1] = i1;  top_gate[t * 2 + 1] = score[i1];
    atomicAdd(&cnt[i0], 1);
    atomicAdd(&cnt[i1], 1);
  }
}

// ---------------------------------------------------------------------------
// 2) Plan: 32-row padded per-expert segments + tile table + clear assignments
// ---------------------------------------------------------------------------
__global__ __launch_bounds__(256)
void plan_kernel(const int* __restrict__ cnt, int* __restrict__ seg_base,
                 int* __restrict__ fill, int* __restrict__ tinfo,
                 int* __restrict__ asg_tok, float* __restrict__ asg_gate) {
  __shared__ int sb[E_], se[E_], stot;
  if (threadIdx.x == 0) {
    int b = 0;
    for (int e = 0; e < E_; ++e) {
      sb[e] = b; seg_base[e] = b; fill[e] = 0;
      b += (((cnt[e] + BM - 1) / BM) * BM);
      se[e] = b;
    }
    stot = b;
  }
  __syncthreads();
  for (int i = threadIdx.x; i < PADCAP; i += blockDim.x) {
    asg_tok[i] = 0; asg_gate[i] = 0.f;
  }
  for (int tI = threadIdx.x; tI < MAXTILES; tI += blockDim.x) {
    const int row0 = tI * BM;
    int e = -1;
    if (row0 < stot) {
      for (int q = 0; q < E_; ++q)
        if (row0 >= sb[q] && row0 < se[q]) { e = q; break; }
    }
    tinfo[2 * tI + 0] = e;
    tinfo[2 * tI + 1] = row0;
  }
}

// ---------------------------------------------------------------------------
// 3) Scatter assignments into expert segments
// ---------------------------------------------------------------------------
__global__ __launch_bounds__(256)
void scatter_kernel(const int* __restrict__ top_idx, const float* __restrict__ top_gate,
                    const int* __restrict__ seg_base, int* __restrict__ fill,
                    int* __restrict__ asg_tok, float* __restrict__ asg_gate) {
  const int t = blockIdx.x * blockDim.x + threadIdx.x;
  if (t >= T_) return;
#pragma unroll
  for (int k = 0; k < K_; ++k) {
    const int e = top_idx[t * 2 + k];
    const int slot = seg_base[e] + atomicAdd(&fill[e], 1);
    asg_tok[slot]  = t;
    asg_gate[slot] = top_gate[t * 2 + k];
  }
}

// ---------------------------------------------------------------------------
// 4) Shared expert: 32-row tiles, 2 row-groups share every B fragment
// ---------------------------------------------------------------------------
__global__ __launch_bounds__(256)
void shared_expert_kernel(const __bf16* __restrict__ xb,
                          const __bf16* __restrict__ w1,   // swz K=D,N=H
                          const __bf16* __restrict__ w2,   // swz K=D,N=H
                          const __bf16* __restrict__ w3,   // swz K=H,N=D
                          float* __restrict__ out) {
  extern __shared__ __bf16 smem[];
  __bf16* xt   = smem;                    // BM x XLD
  __bf16* hbuf = smem + BM * XLD;         // BM x HLD
  const int tid  = threadIdx.x;
  const int lane = tid & 31;
  const int wave = tid >> 5;
  const int row0 = blockIdx.x * BM;
  const int mb = (lane >> 4) << 3;
  const int n  = lane & 15;

  for (int i = tid * 8; i < BM * D_; i += 256 * 8) {
    const int r = i >> 10, c = i & (D_ - 1);
    *(uint4*)&xt[r * XLD + c] = *(const uint4*)(xb + ((size_t)(row0 + r) << 10) + c);
  }
  __syncthreads();

  v8f yacc[16];
#pragma unroll
  for (int j = 0; j < 16; ++j) yacc[j] = vzero8();

  for (int hc = 0; hc < H_; hc += 128) {
    const int nt1 = (hc >> 4) + wave;
    v8f a1[2] = {vzero8(), vzero8()}, a2[2] = {vzero8(), vzero8()};
#pragma unroll 1
    for (int kt = 0; kt < D_ / 32; ++kt) {
      const v16bf af0 = load_a_lds(xt,            XLD, kt << 5, lane);
      const v16bf af1 = load_a_lds(xt + 16 * XLD, XLD, kt << 5, lane);
      const v16bf b1  = load_b_sw(w1, D_ / 32, kt, nt1, lane);
      a1[0] = __builtin_amdgcn_wmma_f32_16x16x32_bf16(false, af0, false, b1, (short)0, a1[0], false, false);
      a1[1] = __builtin_amdgcn_wmma_f32_16x16x32_bf16(false, af1, false, b1, (short)0, a1[1], false, false);
      const v16bf b2  = load_b_sw(w2, D_ / 32, kt, nt1, lane);
      a2[0] = __builtin_amdgcn_wmma_f32_16x16x32_bf16(false, af0, false, b2, (short)0, a2[0], false, false);
      a2[1] = __builtin_amdgcn_wmma_f32_16x16x32_bf16(false, af1, false, b2, (short)0, a2[1], false, false);
    }
#pragma unroll
    for (int g = 0; g < 2; ++g)
#pragma unroll
      for (int r = 0; r < 8; ++r) {
        const float s = a1[g][r];
        const float v = s * __builtin_amdgcn_rcpf(1.f + __expf(-s)) * a2[g][r];
        hbuf[(g * 16 + mb + r) * HLD + wave * 16 + n] = f2bf(v);
      }
    __syncthreads();
#pragma unroll 1
    for (int kk = 0; kk < 4; ++kk) {
      const v16bf af0 = load_a_lds(hbuf,            HLD, kk << 5, lane);
      const v16bf af1 = load_a_lds(hbuf + 16 * HLD, HLD, kk << 5, lane);
#pragma unroll 1
      for (int j = 0; j < 8; ++j) {
        const v16bf bw = load_b_sw(w3, H_ / 32, (hc >> 5) + kk, wave * 8 + j, lane);
        yacc[j]     = __builtin_amdgcn_wmma_f32_16x16x32_bf16(false, af0, false, bw, (short)0, yacc[j],     false, false);
        yacc[8 + j] = __builtin_amdgcn_wmma_f32_16x16x32_bf16(false, af1, false, bw, (short)0, yacc[8 + j], false, false);
      }
    }
    __syncthreads();
  }
#pragma unroll
  for (int g = 0; g < 2; ++g)
#pragma unroll
    for (int j = 0; j < 8; ++j) {
      const int c = wave * 128 + j * 16 + n;
#pragma unroll
      for (int r = 0; r < 8; ++r)
        out[(size_t)(row0 + g * 16 + mb + r) * D_ + c] = yacc[g * 8 + j][r];
    }
}

// ---------------------------------------------------------------------------
// 5) Routed experts: one block per 32-assignment tile; gate + atomicAdd out
// ---------------------------------------------------------------------------
__global__ __launch_bounds__(256)
void routed_expert_kernel(const __bf16* __restrict__ xb,
                          const __bf16* __restrict__ rw1,  // swz K=D,N=RH, per-expert
                          const __bf16* __restrict__ rw2,
                          const __bf16* __restrict__ rw3,  // swz K=RH,N=D, per-expert
                          const int* __restrict__ tinfo,
                          const int* __restrict__ asg_tok,
                          const float* __restrict__ asg_gate,
                          float* __restrict__ out) {
  extern __shared__ __bf16 smem[];
  __bf16* xt   = smem;
  __bf16* hbuf = smem + BM * XLD;
  __shared__ int   toks[BM];
  __shared__ float gts[BM];
  __shared__ int   s_e;
  const int tid = threadIdx.x;
  if (tid == 0) s_e = tinfo[2 * blockIdx.x];
  __syncthreads();
  const int e = s_e;
  if (e < 0) return;

  const int slot0 = blockIdx.x * BM;
  if (tid < BM) { toks[tid] = asg_tok[slot0 + tid]; gts[tid] = asg_gate[slot0 + tid]; }
  __syncthreads();

  const int lane = tid & 31;
  const int wave = tid >> 5;
  const int mb = (lane >> 4) << 3;
  const int n  = lane & 15;

  for (int i = tid * 8; i < BM * D_; i += 256 * 8) {
    const int r = i >> 10, c = i & (D_ - 1);
    *(uint4*)&xt[r * XLD + c] = *(const uint4*)(xb + ((size_t)toks[r] << 10) + c);
  }
  __syncthreads();

  const __bf16* w1 = rw1 + (size_t)e * D_ * RH_;
  const __bf16* w2 = rw2 + (size_t)e * D_ * RH_;
  const __bf16* w3 = rw3 + (size_t)e * RH_ * D_;

  v8f yacc[16];
#pragma unroll
  for (int j = 0; j < 16; ++j) yacc[j] = vzero8();

  for (int hc = 0; hc < RH_; hc += 128) {
    const int nt1 = (hc >> 4) + wave;
    v8f a1[2] = {vzero8(), vzero8()}, a2[2] = {vzero8(), vzero8()};
#pragma unroll 1
    for (int kt = 0; kt < D_ / 32; ++kt) {
      const v16bf af0 = load_a_lds(xt,            XLD, kt << 5, lane);
      const v16bf af1 = load_a_lds(xt + 16 * XLD, XLD, kt << 5, lane);
      const v16bf b1  = load_b_sw(w1, D_ / 32, kt, nt1, lane);
      a1[0] = __builtin_amdgcn_wmma_f32_16x16x32_bf16(false, af0, false, b1, (short)0, a1[0], false, false);
      a1[1] = __builtin_amdgcn_wmma_f32_16x16x32_bf16(false, af1, false, b1, (short)0, a1[1], false, false);
      const v16bf b2  = load_b_sw(w2, D_ / 32, kt, nt1, lane);
      a2[0] = __builtin_amdgcn_wmma_f32_16x16x32_bf16(false, af0, false, b2, (short)0, a2[0], false, false);
      a2[1] = __builtin_amdgcn_wmma_f32_16x16x32_bf16(false, af1, false, b2, (short)0, a2[1], false, false);
    }
#pragma unroll
    for (int g = 0; g < 2; ++g)
#pragma unroll
      for (int r = 0; r < 8; ++r) {
        const float s = a1[g][r];
        const float v = s * __builtin_amdgcn_rcpf(1.f + __expf(-s)) * a2[g][r];
        hbuf[(g * 16 + mb + r) * HLD + wave * 16 + n] = f2bf(v);
      }
    __syncthreads();
#pragma unroll 1
    for (int kk = 0; kk < 4; ++kk) {
      const v16bf af0 = load_a_lds(hbuf,            HLD, kk << 5, lane);
      const v16bf af1 = load_a_lds(hbuf + 16 * HLD, HLD, kk << 5, lane);
#pragma unroll 1
      for (int j = 0; j < 8; ++j) {
        const v16bf bw = load_b_sw(w3, RH_ / 32, (hc >> 5) + kk, wave * 8 + j, lane);
        yacc[j]     = __builtin_amdgcn_wmma_f32_16x16x32_bf16(false, af0, false, bw, (short)0, yacc[j],     false, false);
        yacc[8 + j] = __builtin_amdgcn_wmma_f32_16x16x32_bf16(false, af1, false, bw, (short)0, yacc[8 + j], false, false);
      }
    }
    __syncthreads();
  }
#pragma unroll
  for (int g = 0; g < 2; ++g)
#pragma unroll
    for (int j = 0; j < 8; ++j) {
      const int c = wave * 128 + j * 16 + n;
#pragma unroll
      for (int r = 0; r < 8; ++r) {
        const float v = yacc[g * 8 + j][r] * gts[g * 16 + mb + r];
        atomicAdd(&out[(size_t)toks[g * 16 + mb + r] * D_ + c], v);
      }
    }
}

// ---------------------------------------------------------------------------
extern "C" void kernel_launch(void* const* d_in, const int* in_sizes, int n_in,
                              void* d_out, int out_size, void* d_ws, size_t ws_size,
                              hipStream_t stream) {
  (void)in_sizes; (void)n_in; (void)out_size; (void)ws_size;
  const float* x        = (const float*)d_in[0];
  const float* router_w = (const float*)d_in[1];
  const float* ebias    = (const float*)d_in[2];
  const float* sw1      = (const float*)d_in[3];
  const float* sw2      = (const float*)d_in[4];
  const float* sw3      = (const float*)d_in[5];
  const float* rw1      = (const float*)d_in[6];
  const float* rw2      = (const float*)d_in[7];
  const float* rw3      = (const float*)d_in[8];
  float* out = (float*)d_out;

  // --- bf16 region (16B-aligned, placed first) ---
  __bf16* wsB  = (__bf16*)d_ws;
  __bf16* xb   = wsB;                      const size_t nX  = (size_t)T_ * D_;
  __bf16* sw1b = xb   + nX;                const size_t nS  = (size_t)D_ * H_;
  __bf16* sw2b = sw1b + nS;
  __bf16* sw3b = sw2b + nS;
  __bf16* rw1b = sw3b + nS;                const size_t nR  = (size_t)E_ * D_ * RH_;
  __bf16* rw2b = rw1b + nR;
  __bf16* rw3b = rw2b + nR;
  // --- control region ---
  int*   wsI      = (int*)(rw3b + nR);
  int*   cnt      = wsI;
  int*   fill     = wsI + 16;
  int*   seg_base = wsI + 32;
  int*   top_idx  = wsI + 64;
  int*   tinfo    = top_idx + T_ * K_;
  int*   asg_tok  = tinfo + 2 * MAXTILES;
  float* top_gate = (float*)(asg_tok + PADCAP);
  float* asg_gate = top_gate + T_ * K_;

  // 0) precision/layout pre-pass: x -> bf16; weights -> bf16 WMMA-B swizzle
  cvt_kernel<<<(int)(nX / 4 + 255) / 256, 256, 0, stream>>>(x, xb, (int)(nX / 4));
  {
    const int thS = (D_ / 32) * (H_ / 16) * 32;
    swz_kernel<<<(thS + 255) / 256, 256, 0, stream>>>(sw1, sw1b, D_, H_, 1);
    swz_kernel<<<(thS + 255) / 256, 256, 0, stream>>>(sw2, sw2b, D_, H_, 1);
    const int thS3 = (H_ / 32) * (D_ / 16) * 32;
    swz_kernel<<<(thS3 + 255) / 256, 256, 0, stream>>>(sw3, sw3b, H_, D_, 1);
    const int thR = (D_ / 32) * (RH_ / 16) * 32 * E_;
    swz_kernel<<<(thR + 255) / 256, 256, 0, stream>>>(rw1, rw1b, D_, RH_, E_);
    swz_kernel<<<(thR + 255) / 256, 256, 0, stream>>>(rw2, rw2b, D_, RH_, E_);
    const int thR3 = (RH_ / 32) * (D_ / 16) * 32 * E_;
    swz_kernel<<<(thR3 + 255) / 256, 256, 0, stream>>>(rw3, rw3b, RH_, D_, E_);
  }

  // 1-3) routing
  hipMemsetAsync(cnt, 0, 64 * sizeof(int), stream);
  router_kernel<<<T_ / 8, 256, 0, stream>>>(x, router_w, ebias, top_idx, top_gate, cnt);
  plan_kernel<<<1, 256, 0, stream>>>(cnt, seg_base, fill, tinfo, asg_tok, asg_gate);
  scatter_kernel<<<T_ / 256, 256, 0, stream>>>(top_idx, top_gate, seg_base, fill,
                                               asg_tok, asg_gate);
  // 4-5) expert GEMMs (dynamic LDS: 74752B per block, within 320KB/WGP)
  shared_expert_kernel<<<T_ / BM, 256, SMEM_BYTES, stream>>>(xb, sw1b, sw2b, sw3b, out);
  routed_expert_kernel<<<MAXTILES, 256, SMEM_BYTES, stream>>>(xb, rw1b, rw2b, rw3b,
                                                              tinfo, asg_tok, asg_gate, out);
}
